// ClimaXEmbedding_77515569758478
// MI455X (gfx1250) — matrix operand order repaired and checked by reading, not compiled, not run on hardware
//
#include <hip/hip_runtime.h>

// ---------------------------------------------------------------------------
// ClimaX embedding, fused for MI455X (gfx1250, wave32, WMMA bf16 16x16x32).
// Algebra: scores = (fold q into wk) U @ tok^T ; ctx = wv @ (attn-mixed tok) + bv
// so the big per-channel K/V projections (274 GFLOP) collapse to ~24 GFLOP.
// All intermediates (tok tile, scores, attn, m, ctx) stay in LDS.
// ---------------------------------------------------------------------------

typedef __attribute__((ext_vector_type(16))) __bf16 v16bf;
typedef __attribute__((ext_vector_type(8)))  __bf16 v8bf;
typedef __attribute__((ext_vector_type(8)))  float  v8f;

#define WMMA_BF16(A, B, C) \
  __builtin_amdgcn_wmma_f32_16x16x32_bf16(false, (A), false, (B), (short)0, (C), false, false)

__device__ __forceinline__ v8f zero_v8f() {
  v8f z;
#pragma unroll
  for (int i = 0; i < 8; ++i) z[i] = 0.0f;
  return z;
}
__device__ __forceinline__ v16bf zero_v16bf() {
  v16bf z;
#pragma unroll
  for (int i = 0; i < 16; ++i) z[i] = (__bf16)0.0f;
  return z;
}

// A fragment 16x32 bf16 from row-major [row][col] (stride rs elems).
// Layout: elem e -> K = half*16+e (contiguous).
__device__ __forceinline__ v16bf load_a32(const __bf16* base, int row, int rs, int cb, int hl) {
  const __bf16* p = base + row * rs + cb + hl * 16;
  return *(const v16bf*)p;
}

// A fragment with only K=0..15 real (K>=16 zero-padded).
__device__ __forceinline__ v16bf load_a16(const __bf16* base, int row, int rs, int hl) {
  const __bf16* p = base + row * rs;
  if (hl == 0) return *(const v16bf*)p;
  return zero_v16bf();
}

// ---------------------------------------------------------------------------
// Precompute kernels
// ---------------------------------------------------------------------------
__global__ void cvt_bf16_k(const float* __restrict__ s, __bf16* __restrict__ d, int n) {
  int i = blockIdx.x * blockDim.x + threadIdx.x;
  if (i < n) d[i] = (__bf16)s[i];
}
__global__ void add2_k(const float* __restrict__ a, const float* __restrict__ b,
                       float* __restrict__ d, int n) {
  int i = blockIdx.x * blockDim.x + threadIdx.x;
  if (i < n) d[i] = a[i] + b[i];
}
// q[e] = sum_d wq[e,d]*cq[d] + bq[e]
__global__ void calc_q_k(const float* __restrict__ wq, const float* __restrict__ bq,
                         const float* __restrict__ cq, float* __restrict__ q) {
  int e = blockIdx.x * blockDim.x + threadIdx.x;
  if (e >= 1024) return;
  float s = bq[e];
  const float* row = wq + e * 1024;
  for (int d = 0; d < 1024; ++d) s += row[d] * cq[d];
  q[e] = s;
}
// U[h,d] = 0.125 * sum_j q[h*64+j] * wk[h*64+j, d]   (bk drops out of softmax)
__global__ void calc_U_k(const float* __restrict__ q, const float* __restrict__ wk,
                         __bf16* __restrict__ U) {
  int idx = blockIdx.x * blockDim.x + threadIdx.x;
  if (idx >= 16 * 1024) return;
  int h = idx >> 10, d = idx & 1023;
  float s = 0.0f;
  for (int j = 0; j < 64; ++j) s += q[h * 64 + j] * wk[(h * 64 + j) * 1024 + d];
  U[idx] = (__bf16)(s * 0.125f);
}

// ---------------------------------------------------------------------------
// Fused main kernel: one WG (4 waves) per 16-token tile. 256 WGs total.
// ---------------------------------------------------------------------------
__global__ __launch_bounds__(128, 1) void climax_fused(
    const float*  __restrict__ x,       // [B,16,128,256]
    const __bf16* __restrict__ pw,      // [16][1024][16]  proj_w bf16
    const float*  __restrict__ biasvd,  // [16][1024] proj_b + channel_embed
    const float*  __restrict__ pos,     // [2048][1024]
    const __bf16* __restrict__ U,       // [16][1024]
    const __bf16* __restrict__ wvb,     // [1024][1024] (e,d)
    const float*  __restrict__ bv,      // [1024]
    const __bf16* __restrict__ wob,     // [1024][1024] (f,e)
    const float*  __restrict__ bo,      // [1024]
    float*        __restrict__ out)     // [B,2048,1024]
{
  constexpr int CH = 64;   // d-chunk
  constexpr int NCH = 16;  // 1024 / 64

  __shared__ __align__(64) __bf16 paL[16][16][16];   // patches  [v][t][k]   8KB
  __shared__ __align__(64) __bf16 tokA[16][16][CH];  // tok      [t][v][d]  32KB
  __shared__ __align__(64) __bf16 tokT[16][CH][16];  // tok^T    [t][d][v]  32KB
  __shared__ __align__(64) __bf16 mL[16][16][CH];    // mixed    [h][t][d]  32KB
  __shared__ __align__(64) __bf16 atL[16][16][16];   // attn     [t][h][v]   8KB
  __shared__ __align__(64) char   blob[16 * 1024 * 2];  // 32KB, reused:
  float (*scL)[16][16] = (float(*)[16][16])blob;        //   scores [t][h][v] (16KB)
  __bf16 (*ctxL)[1024] = (__bf16(*)[1024])blob;         //   ctx    [t][e]    (32KB)

  const int b    = blockIdx.x >> 7;          // 128 tiles per batch
  const int l0   = (blockIdx.x & 127) * 16;  // first token of tile
  const int wave = threadIdx.x >> 5;
  const int lane = threadIdx.x & 31;
  const int lr   = lane & 15;
  const int hl   = lane >> 4;
  const int wv4  = wave * 4;

  // ---- stage patch tile: x[b,v, hp*4+py, wp*4+px] -> paL[v][t][k] (bf16) ----
  for (int i = threadIdx.x; i < 16 * 16 * 16; i += 128) {
    int k = i & 15, t = (i >> 4) & 15, v = i >> 8;
    int l = l0 + t, hp = l >> 6, wp = l & 63, py = k >> 2, px = k & 3;
    paL[v][t][k] = (__bf16)x[(((b * 16) + v) * 128 + hp * 4 + py) * 256 + wp * 4 + px];
  }
  __syncthreads();

  // =====================  PASS 1: tok chunks -> scores  =====================
  v8f scacc[4];
#pragma unroll
  for (int i = 0; i < 4; ++i) scacc[i] = zero_v8f();

  for (int ch = 0; ch < NCH; ++ch) {
    const int d0 = ch * CH;
    // -- tok phase: each wave owns 4 variables --
#pragma unroll
    for (int i = 0; i < 4; ++i) {
      const int v = wv4 + i;
      v16bf ap = load_a16(&paL[v][0][0], lr, 16, hl);  // A: patches 16t x 16k(pad32)
#pragma unroll
      for (int c = 0; c < 4; ++c) {
        const int dn = d0 + c * 16 + lr;  // this lane's output column (global d)
        v16bf bw;
        if (hl == 0) bw = *(const v16bf*)(pw + ((v << 10) + dn) * 16);  // B: pw^T, K=k
        else         bw = zero_v16bf();                                  // pad K>=16
        v8f acc = zero_v8f();
        acc = WMMA_BF16(ap, bw, acc);
        const float bias = biasvd[(v << 10) + dn];
#pragma unroll
        for (int r = 0; r < 8; ++r) {
          const int t = hl * 8 + r;
          __bf16 bb = (__bf16)(acc[r] + bias + pos[(l0 + t) * 1024 + dn]);
          tokA[t][v][c * 16 + lr] = bb;
          tokT[t][c * 16 + lr][v] = bb;
        }
      }
    }
    __syncthreads();
    // -- scores phase: each wave owns 4 tokens: scores_t += U @ tok_t^T --
#pragma unroll
    for (int i = 0; i < 4; ++i) {
      const int t = wv4 + i;
#pragma unroll
      for (int s = 0; s < 2; ++s) {
        v16bf aU = load_a32(U, lr, 1024, d0 + s * 32, hl);                  // A: U 16h x 32d
        v16bf bT = *(const v16bf*)(&tokA[t][lr][s * 32 + hl * 16]);         // B: K=d, N=v
        scacc[i] = WMMA_BF16(aU, bT, scacc[i]);
      }
    }
    __syncthreads();
  }

  // ---- dump scores, softmax over v (16 wide) ----
#pragma unroll
  for (int i = 0; i < 4; ++i) {
    const int t = wv4 + i;
#pragma unroll
    for (int r = 0; r < 8; ++r) scL[t][hl * 8 + r][lr] = scacc[i][r];
  }
  __syncthreads();
  for (int rr = threadIdx.x; rr < 256; rr += 128) {
    const int t = rr >> 4, h = rr & 15;
    float mx = -3.0e38f;
#pragma unroll
    for (int v = 0; v < 16; ++v) mx = fmaxf(mx, scL[t][h][v]);
    float ex[16], sum = 0.0f;
#pragma unroll
    for (int v = 0; v < 16; ++v) { ex[v] = __expf(scL[t][h][v] - mx); sum += ex[v]; }
    const float inv = 1.0f / sum;
#pragma unroll
    for (int v = 0; v < 16; ++v) atL[t][h][v] = (__bf16)(ex[v] * inv);
  }
  __syncthreads();

  // =============  PASS 2: recompute tok -> mix -> ctx accumulate  ===========
  v8f cacc[16];
#pragma unroll
  for (int i = 0; i < 16; ++i) cacc[i] = zero_v8f();

  for (int ch = 0; ch < NCH; ++ch) {
    const int d0 = ch * CH;
    // -- tok recompute (patch embed is cheap: K=16) --
#pragma unroll
    for (int i = 0; i < 4; ++i) {
      const int v = wv4 + i;
      v16bf ap = load_a16(&paL[v][0][0], lr, 16, hl);
#pragma unroll
      for (int c = 0; c < 4; ++c) {
        const int dn = d0 + c * 16 + lr;
        v16bf bw;
        if (hl == 0) bw = *(const v16bf*)(pw + ((v << 10) + dn) * 16);
        else         bw = zero_v16bf();
        v8f acc = zero_v8f();
        acc = WMMA_BF16(ap, bw, acc);
        const float bias = biasvd[(v << 10) + dn];
#pragma unroll
        for (int r = 0; r < 8; ++r) {
          const int t = hl * 8 + r;
          __bf16 bb = (__bf16)(acc[r] + bias + pos[(l0 + t) * 1024 + dn]);
          tokA[t][v][c * 16 + lr] = bb;
          tokT[t][c * 16 + lr][v] = bb;
        }
      }
    }
    __syncthreads();
    // -- mix phase: m_t = attn_t @ tok_t (each wave owns 4 tokens) --
#pragma unroll
    for (int i = 0; i < 4; ++i) {
      const int t = wv4 + i;
      v16bf aAt = load_a16(&atL[t][0][0], lr, 16, hl);  // A: attn 16h x 16v(pad32)
#pragma unroll
      for (int c = 0; c < 4; ++c) {
        v16bf bm;
        if (hl == 0) bm = *(const v16bf*)(&tokT[t][c * 16 + lr][0]);  // B: K=v, N=d
        else         bm = zero_v16bf();
        v8f macc = zero_v8f();
        macc = WMMA_BF16(aAt, bm, macc);
#pragma unroll
        for (int r = 0; r < 8; ++r)
          mL[hl * 8 + r][t][c * 16 + lr] = (__bf16)macc[r];
      }
    }
    __syncthreads();
    // -- ctx phase: ctx_h += m_h @ wv_h^T (each wave owns 4 heads) --
#pragma unroll
    for (int i = 0; i < 4; ++i) {
      const int h = wv4 + i;
      __builtin_prefetch(wvb + (h * 64 + lr) * 1024 + d0 + CH, 0, 1);
#pragma unroll
      for (int s = 0; s < 2; ++s) {
        v16bf aM = load_a32(&mL[h][0][0], lr, CH, s * 32, hl);  // A: m 16t x 32d
#pragma unroll
        for (int jt = 0; jt < 4; ++jt) {
          const int e = h * 64 + jt * 16 + lr;
          v16bf bW = *(const v16bf*)(wvb + e * 1024 + d0 + s * 32 + hl * 16);  // B: K=d,N=j
          cacc[i * 4 + jt] = WMMA_BF16(aM, bW, cacc[i * 4 + jt]);
        }
      }
    }
    __syncthreads();
  }

  // ---- ctx -> LDS (bf16), add bv ----
#pragma unroll
  for (int i = 0; i < 4; ++i) {
    const int h = wv4 + i;
#pragma unroll
    for (int jt = 0; jt < 4; ++jt) {
      const int e = h * 64 + jt * 16 + lr;
      const float bve = bv[e];
#pragma unroll
      for (int r = 0; r < 8; ++r)
        ctxL[hl * 8 + r][e] = (__bf16)(cacc[i * 4 + jt][r] + bve);
    }
  }
  __syncthreads();

  // =====================  OUT: out = ctx @ wo^T + bo  =======================
  v8f oacc[16];
#pragma unroll
  for (int j = 0; j < 16; ++j) oacc[j] = zero_v8f();

  for (int ks = 0; ks < 32; ++ks) {
    v16bf aC = load_a32(&ctxL[0][0], lr, 1024, ks * 32, hl);  // A: ctx 16t x 32e
    __builtin_prefetch(wob + (wave * 256 + lr) * 1024 + ks * 32 + 32, 0, 1);
#pragma unroll
    for (int j = 0; j < 16; ++j) {
      const int f = (wave * 16 + j) * 16 + lr;
      v16bf bO = *(const v16bf*)(wob + f * 1024 + ks * 32 + hl * 16);  // B: K=e, N=f
      oacc[j] = WMMA_BF16(aC, bO, oacc[j]);
    }
  }
#pragma unroll
  for (int j = 0; j < 16; ++j) {
    const int f = (wave * 16 + j) * 16 + lr;
    const float bof = bo[f];
#pragma unroll
    for (int r = 0; r < 8; ++r)
      out[((size_t)b * 2048 + l0 + hl * 8 + r) * 1024 + f] = oacc[j][r] + bof;
  }
}

// ---------------------------------------------------------------------------
extern "C" void kernel_launch(void* const* d_in, const int* in_sizes, int n_in,
                              void* d_out, int out_size, void* d_ws, size_t ws_size,
                              hipStream_t stream) {
  const float* x    = (const float*)d_in[0];
  const float* pjw  = (const float*)d_in[1];   // [16,1024,16]
  const float* pjb  = (const float*)d_in[2];   // [16,1024]
  const float* cemb = (const float*)d_in[3];   // [1,16,1024]
  const float* pos  = (const float*)d_in[4];   // [1,2048,1024]
  const float* cq   = (const float*)d_in[5];   // [1,1,1024]
  const float* wq   = (const float*)d_in[6];
  const float* wk   = (const float*)d_in[7];
  const float* wv   = (const float*)d_in[8];
  const float* bq   = (const float*)d_in[9];
  // d_in[10] = bk : drops out of softmax (per-row constant)
  const float* bv   = (const float*)d_in[11];
  const float* wo   = (const float*)d_in[12];
  const float* bo   = (const float*)d_in[13];
  float* out = (float*)d_out;

  char* ws = (char*)d_ws;
  __bf16* pwbf   = (__bf16*)(ws);                       // 512KB
  __bf16* wvbf   = (__bf16*)(ws + (512u << 10));        // 2MB
  __bf16* wobf   = (__bf16*)(ws + (2560u << 10));       // 2MB
  float*  biasvd = (float*) (ws + (4608u << 10));       // 64KB
  float*  qv     = (float*) (ws + (4672u << 10));       // 4KB
  __bf16* Ubf    = (__bf16*)(ws + (4676u << 10));       // 32KB

  cvt_bf16_k<<<1024, 256, 0, stream>>>(pjw, pwbf, 16 * 1024 * 16);
  cvt_bf16_k<<<4096, 256, 0, stream>>>(wv, wvbf, 1024 * 1024);
  cvt_bf16_k<<<4096, 256, 0, stream>>>(wo, wobf, 1024 * 1024);
  add2_k<<<64, 256, 0, stream>>>(pjb, cemb, biasvd, 16 * 1024);
  calc_q_k<<<4, 256, 0, stream>>>(wq, bq, cq, qv);
  calc_U_k<<<64, 256, 0, stream>>>(qv, wk, Ubf);

  climax_fused<<<256, 128, 0, stream>>>(x, pwbf, biasvd, pos, Ubf,
                                        wvbf, bv, wobf, bo, out);
}
